// ResidualVectorQuantizer_29592324669664
// MI455X (gfx1250) — matrix operand order repaired and checked by reading, not compiled, not run on hardware
//
#include <hip/hip_runtime.h>

typedef __bf16 bf16;
typedef __attribute__((ext_vector_type(16))) __bf16 v16bf;
typedef __attribute__((ext_vector_type(8)))  float  v8f;

#define T_TOT   65536
#define D_DIM   512
#define NQ      8
#define K_CB    1024
#define TILE_T  64
#define NWAVES  8
#define NT_PER_WAVE 8            // 1024 cols / 16 / 8 waves
#define KSTEPS  (D_DIM / 32)     // 16 wmma K-steps over D

// -------- prep 1: ||c||^2 per codeword (one wave per codeword) --------
__global__ void rvq_prep_csq(const float* __restrict__ cb, float* __restrict__ csq) {
  int wave = (blockIdx.x * blockDim.x + threadIdx.x) >> 5;
  int lane = threadIdx.x & 31;
  if (wave >= NQ * K_CB) return;
  const float* row = cb + (size_t)wave * D_DIM;
  float s = 0.f;
#pragma unroll
  for (int i = 0; i < D_DIM / 32; ++i) {
    float x = row[lane + i * 32];
    s += x * x;
  }
#pragma unroll
  for (int m = 16; m >= 1; m >>= 1) s += __shfl_xor(s, m, 32);
  if (lane == 0) csq[wave] = s;
}

// -------- prep 2: codebook -> WMMA-B-fragment-ready bf16 layout --------
// B is 32x16 bf16 per fragment: lanes 0-15 hold K(d)=0..15 of column N=lane,
// lanes 16-31 hold K(d)=16..31 of column N=lane-16 (16 contiguous bf16/lane).
__global__ void rvq_prep_frags(const float* __restrict__ cb, bf16* __restrict__ frags) {
  int gid = blockIdx.x * blockDim.x + threadIdx.x;   // one per lane-fragment
  if (gid >= NQ * 64 * KSTEPS * 32) return;
  int lane = gid & 31;
  int ks   = (gid >> 5) & (KSTEPS - 1);
  int nt   = (gid >> 9) & 63;
  int q    = gid >> 15;
  int kcol  = nt * 16 + (lane & 15);
  int dbase = ks * 32 + ((lane < 16) ? 0 : 16);
  const float* src = cb + ((size_t)(q * K_CB + kcol)) * D_DIM + dbase;
  bf16* dst = frags + (size_t)gid * 16;
#pragma unroll
  for (int s = 0; s < 16; ++s) dst[s] = (bf16)src[s];
}

// -------- main: per-block 64-token tile, all 8 RVQ levels in LDS --------
__global__ __launch_bounds__(256, 1)
void rvq_main(const float* __restrict__ hs, const float* __restrict__ cb,
              const float* __restrict__ csq, const bf16* __restrict__ frags,
              int* __restrict__ codes_out, float* __restrict__ quant_out) {
  extern __shared__ __align__(16) char smem[];
  float* resid = (float*)smem;                                   // 64*512 f32  (128 KB)
  bf16*  afrag = (bf16*)(smem + TILE_T * D_DIM * 4);             // 4*16*32*16  ( 64 KB)
  unsigned long long* best =
      (unsigned long long*)(smem + TILE_T * D_DIM * 4 + 4 * KSTEPS * 32 * 16 * 2);
  int* sidx = (int*)(best + TILE_T);

  const int tid  = threadIdx.x;
  const int lane = tid & 31;
  const int w    = tid >> 5;
  const int t0   = blockIdx.x * TILE_T;

  // stage hidden tile -> resid (f32)
#pragma unroll
  for (int i = 0; i < (TILE_T * D_DIM) / (256 * 4); ++i) {
    int e = (i * 256 + tid) * 4;
    *(float4*)(resid + e) = *(const float4*)(hs + (size_t)t0 * D_DIM + e);
  }
  __syncthreads();

#pragma unroll 1
  for (int q = 0; q < NQ; ++q) {
    if (tid < TILE_T) best[tid] = ~0ull;

    // re-swizzle residual f32 -> bf16 A fragments (16-bit A 16x32 lane layout)
#pragma unroll 8
    for (int i = 0; i < (4 * KSTEPS * 32 * 16) / 256; ++i) {
      int e  = tid * 128 + i;
      int s  = e & 15;
      int ln = (e >> 4) & 31;
      int ks = (e >> 9) & 15;
      int rt = e >> 13;
      int m  = rt * 16 + (ln & 15);
      int off = ((ln < 16) ? 0 : 16) + s;
      afrag[e] = (bf16)resid[m * D_DIM + ks * 32 + off];
    }
    __syncthreads();

    // running per-lane argmin state for all 4 row-tiles (kept in registers:
    // every index below is a compile-time constant after unrolling)
    float bv[4][8]; int bk[4][8];
#pragma unroll
    for (int rt = 0; rt < 4; ++rt)
#pragma unroll
      for (int j = 0; j < 8; ++j) { bv[rt][j] = 3.4e38f; bk[rt][j] = 0x7fffffff; }

    const bf16* ab = afrag + lane * 16;   // A fragment base for this lane

#pragma unroll 1   // keep B loads inside: one load per fragment, no hoist-spill
    for (int nt = 0; nt < NT_PER_WAVE; ++nt) {
      const bf16* bp =
          frags + ((size_t)((q * 64 + w * NT_PER_WAVE + nt) * KSTEPS) * 32 + lane) * 16;
      __builtin_prefetch(bp, 0, 3);

      v8f acc[4];
#pragma unroll
      for (int rt = 0; rt < 4; ++rt) acc[rt] = (v8f){};

#pragma unroll 4
      for (int ks = 0; ks < KSTEPS; ++ks) {
        v16bf b = *(const v16bf*)(bp + ks * (32 * 16));   // 1 global load, used 4x
#pragma unroll
        for (int rt = 0; rt < 4; ++rt) {
          v16bf a = *(const v16bf*)(ab + (rt * KSTEPS + ks) * (32 * 16));
          acc[rt] = __builtin_amdgcn_wmma_f32_16x16x32_bf16(
              false, a, false, b, (short)0, acc[rt], false, false);
        }
      }

      int   kcol = (w * NT_PER_WAVE + nt) * 16 + (lane & 15);
      float csqn = csq[q * K_CB + kcol];     // single scalar load per nt
#pragma unroll
      for (int rt = 0; rt < 4; ++rt)
#pragma unroll
        for (int j = 0; j < 8; ++j) {
          float sc = csqn - 2.0f * acc[rt][j];   // ||r||^2 constant per row: dropped
          if (sc < bv[rt][j] || (sc == bv[rt][j] && kcol < bk[rt][j])) {
            bv[rt][j] = sc; bk[rt][j] = kcol;
          }
        }
    }

    // 16-lane butterfly argmin per row, then one ds_min_u64 per row
#pragma unroll
    for (int rt = 0; rt < 4; ++rt)
#pragma unroll
      for (int j = 0; j < 8; ++j) {
        float v = bv[rt][j]; int k = bk[rt][j];
#pragma unroll
        for (int m = 8; m >= 1; m >>= 1) {
          float ov = __shfl_xor(v, m, 32);
          int   ok = __shfl_xor(k, m, 32);
          if (ov < v || (ov == v && ok < k)) { v = ov; k = ok; }
        }
        if ((lane & 15) == 0) {
          unsigned u = __float_as_uint(v);
          unsigned key = (u & 0x80000000u) ? ~u : (u ^ 0x80000000u);  // order-preserving
          unsigned long long pk = ((unsigned long long)key << 32) | (unsigned)k;
          int row = rt * 16 + ((lane < 16) ? j : j + 8);
          atomicMin(&best[row], pk);
        }
      }
    __syncthreads();

    if (tid < TILE_T) {
      int k = (int)(best[tid] & 0xffffffffu);
      sidx[tid] = k;
      codes_out[(size_t)q * T_TOT + t0 + tid] = k;
    }
    __syncthreads();

    // residual update: r -= cb[q][idx]  (4 threads per token row)
    {
      int m = tid >> 2, sub = tid & 3;
      const float* cw = cb + ((size_t)(q * K_CB + sidx[m])) * D_DIM;
#pragma unroll
      for (int i = 0; i < 32; ++i) {
        int d = i * 16 + sub * 4;
        float4 c4 = *(const float4*)(cw + d);
        float4 r4 = *(float4*)(resid + m * D_DIM + d);
        r4.x -= c4.x; r4.y -= c4.y; r4.z -= c4.z; r4.w -= c4.w;
        *(float4*)(resid + m * D_DIM + d) = r4;
      }
    }
    __syncthreads();
  }

  // quantized = hidden - final residual
#pragma unroll
  for (int i = 0; i < (TILE_T * D_DIM) / (256 * 4); ++i) {
    int e = (i * 256 + tid) * 4;
    float4 h = *(const float4*)(hs + (size_t)t0 * D_DIM + e);
    float4 r = *(float4*)(resid + e);
    float4 o = make_float4(h.x - r.x, h.y - r.y, h.z - r.z, h.w - r.w);
    *(float4*)(quant_out + (size_t)t0 * D_DIM + e) = o;
  }
}

extern "C" void kernel_launch(void* const* d_in, const int* in_sizes, int n_in,
                              void* d_out, int out_size, void* d_ws, size_t ws_size,
                              hipStream_t stream) {
  const float* hs = (const float*)d_in[0];
  const float* cb = (const float*)d_in[1];

  float* csq   = (float*)d_ws;                                        // 32 KB
  bf16*  frags = (bf16*)((char*)d_ws + (size_t)NQ * K_CB * sizeof(float)); // 8 MB

  int*   codes = (int*)d_out;                                         // NQ*T int32 (bitcast)
  float* quant = (float*)d_out + (size_t)NQ * T_TOT;                  // T*D f32

  rvq_prep_csq<<<dim3((NQ * K_CB) / 8), dim3(256), 0, stream>>>(cb, csq);
  rvq_prep_frags<<<dim3((NQ * 64 * KSTEPS * 32) / 256), dim3(256), 0, stream>>>(cb, frags);

  size_t smem = (size_t)TILE_T * D_DIM * 4       // resid f32
              + (size_t)4 * KSTEPS * 32 * 16 * 2 // afrag bf16
              + (size_t)TILE_T * 8               // best u64
              + (size_t)TILE_T * 4;              // sidx
  rvq_main<<<dim3(T_TOT / TILE_T), dim3(256), smem, stream>>>(
      hs, cb, csq, frags, codes, quant);
}
